// TransformerLayer_46351287058751
// MI455X (gfx1250) — matrix operand order repaired and checked by reading, not compiled, not run on hardware
//
#include <hip/hip_runtime.h>
#include <stdint.h>

#define BATCH 2048
#define LSEQ  128
#define DIM   512
#define STR   520          // padded LDS row stride in bf16 elements (1040 B, 16B aligned, bank-skewed)
#define NSLOPE 0.2f
#define WMAT  (DIM * DIM)  // elements per weight matrix

typedef __attribute__((ext_vector_type(16))) __bf16 bf16x16;
typedef __attribute__((ext_vector_type(8)))  __bf16 bf16x8;
typedef __attribute__((ext_vector_type(8)))  float  f32x8;

union AFrag  { uint32_t u[8]; bf16x16 v; };
union Pack8  { bf16x8 v; uint4 u4; };
union BF16U  { __bf16 h; unsigned short s; };

__device__ __forceinline__ unsigned short f2bfh(float f) {
  BF16U u; u.h = (__bf16)f; return u.s;
}
__device__ __forceinline__ float bf2f(unsigned short h) {
  return __uint_as_float(((uint32_t)h) << 16);
}
__device__ __forceinline__ float lrelu(float v) { return v > 0.f ? v : NSLOPE * v; }

// ---------------- pre-pass: fp32 weights -> bf16 (runs once per launch) ----------------
__global__ __launch_bounds__(256)
void cvt_weights_kernel(const float* __restrict__ src, unsigned short* __restrict__ dst) {
  int i = (blockIdx.x * 256 + threadIdx.x) * 8;   // 256K elems / 8 = 32768 threads = 128 blocks
  union { float4 q4[2]; f32x8 v; } raw;
  const float4* p = (const float4*)(src + i);
  raw.q4[0] = p[0]; raw.q4[1] = p[1];
  Pack8 pk; pk.v = __builtin_convertvector(raw.v, bf16x8);
  *(uint4*)(dst + i) = pk.u4;
}

// A fragment (16x32 bf16, ISA layout) from LDS row-major tile, stride STR.
// lane l%16 -> row; lanes 0-15 hold K=[0..7]+[16..23], lanes 16-31 hold K=[8..15]+[24..31]
__device__ __forceinline__ bf16x16 load_a(const unsigned short* base, int mbase, int k0, int lane) {
  int m = lane & 15, hi = lane >> 4;
  const unsigned short* p = base + (size_t)(mbase + m) * STR + k0 + hi * 8;
  uint4 c1 = *(const uint4*)p;
  uint4 c2 = *(const uint4*)(p + 16);
  AFrag r;
  r.u[0] = c1.x; r.u[1] = c1.y; r.u[2] = c1.z; r.u[3] = c1.w;
  r.u[4] = c2.x; r.u[5] = c2.y; r.u[6] = c2.z; r.u[7] = c2.w;
  return r.v;
}

// B fragment (32x16) from pre-converted bf16 weights: B[k][n] = Wb[n0+n][k0+k].
// lane%16 -> output column (row of Wb); lanes 0-15 hold K=[0..7]+[16..23], lanes 16-31 hold K=[8..15]+[24..31] (2x b128).
__device__ __forceinline__ bf16x16 load_b_bf(const unsigned short* Wb, int n0, int k0, int lane) {
  const unsigned short* p = Wb + (size_t)(n0 + (lane & 15)) * DIM + k0 + ((lane >> 4) << 3);
  uint4 c1 = *(const uint4*)p;
  uint4 c2 = *(const uint4*)(p + 16);
  AFrag r;
  r.u[0] = c1.x; r.u[1] = c1.y; r.u[2] = c1.z; r.u[3] = c1.w;
  r.u[4] = c2.x; r.u[5] = c2.y; r.u[6] = c2.z; r.u[7] = c2.w;
  return r.v;
}

// 8 M-tiles x 4 N-strips for one K step: each A fragment feeds 4 back-to-back WMMAs.
__device__ __forceinline__ void wmma_block(const unsigned short* Albs, int k0, int lane,
                                           const bf16x16 bf[4], f32x8 acc[4][8]) {
#pragma unroll
  for (int m = 0; m < 8; ++m) {
    bf16x16 af = load_a(Albs, m << 4, k0, lane);
#pragma unroll
    for (int s = 0; s < 4; ++s)
      acc[s][m] = __builtin_amdgcn_wmma_f32_16x16x32_bf16(false, af, false, bf[s],
                                                          (short)0, acc[s][m], false, false);
  }
}

// Four adjacent 16-wide N strips (a wave's full private 64 columns), 8 M-tiles x K=512.
// B fragments ping-pong between two register buffers (no copies, no WMMA->VALU hazard movs);
// next K-step's bf16 B loads issue before the current WMMA chain (L2-hot).
__device__ __forceinline__ void gemm_strip4(const unsigned short* Albs, const unsigned short* Wb,
                                            int n0, int lane, f32x8 acc[4][8]) {
  bf16x16 b0[4], b1[4];
#pragma unroll
  for (int s = 0; s < 4; ++s) b0[s] = load_b_bf(Wb, n0 + 16 * s, 0, lane);
#pragma unroll 1
  for (int kt = 0; kt < 16; kt += 2) {
    int k0 = kt << 5;
#pragma unroll
    for (int s = 0; s < 4; ++s) b1[s] = load_b_bf(Wb, n0 + 16 * s, k0 + 32, lane);
    wmma_block(Albs, k0, lane, b0, acc);
    if (kt < 14) {
#pragma unroll
      for (int s = 0; s < 4; ++s) b0[s] = load_b_bf(Wb, n0 + 16 * s, k0 + 64, lane);
    }
    wmma_block(Albs, k0 + 32, lane, b1, acc);
  }
}

// ---- epilogues (per 16-wide strip) ----
__device__ __forceinline__ void epi_yh(const f32x8 acc[8], unsigned short* yhB,
                                       const float* fcbL, int n0, int lane) {
  int col = n0 + (lane & 15), hi = lane >> 4;
  float bias = fcbL[col];
#pragma unroll
  for (int m = 0; m < 8; ++m)
#pragma unroll
    for (int r = 0; r < 8; ++r) {
      float v = lrelu(acc[m][r] + bias);
      yhB[(size_t)(m * 16 + r + 8 * hi) * STR + col] = f2bfh(v);
    }
}

__device__ __forceinline__ void epi_att(const f32x8 acc[8], unsigned short* attB,
                                        const float* kbL, const float* qL, int n0, int lane) {
  int col = n0 + (lane & 15), hi = lane >> 4;
  float kbv = kbL[col], qv = qL[col];
#pragma unroll
  for (int m = 0; m < 8; ++m)
#pragma unroll
    for (int r = 0; r < 8; ++r) {
      float att = (acc[m][r] + kbv) * qv;
      attB[(size_t)(m * 16 + r + 8 * hi) * STR + col] = f2bfh(att);
    }
}

__device__ __forceinline__ void epi_vec(const f32x8 acc[8], const unsigned short* attB,
                                        const float* vbL, const float* izL, float* hsL,
                                        int n0, int lane) {
  int col = n0 + (lane & 15), hi = lane >> 4;
  float vbv = vbL[col], iz = izL[col];
  float part = 0.f;
#pragma unroll
  for (int m = 0; m < 8; ++m)
#pragma unroll
    for (int r = 0; r < 8; ++r) {
      int l = m * 16 + r + 8 * hi;
      float e = bf2f(attB[(size_t)l * STR + col]) * iz;
      float t = (acc[m][r] + vbv) * e;
      part += lrelu(t);
    }
  atomicAdd(&hsL[col], part);   // ds_add_f32; only 2 lanes collide per address
}

__global__ __launch_bounds__(256)
void ext_attn_kernel(const float* __restrict__ x,   const float* __restrict__ y,
                     const float* __restrict__ fc_w, const float* __restrict__ fc_b,
                     const float* __restrict__ q_w,  const float* __restrict__ q_b,
                     const float* __restrict__ k_b,  const float* __restrict__ v_b,
                     const unsigned short* __restrict__ wbf,   // bf16 fc_w | k_w | v_w
                     float* __restrict__ out) {
  extern __shared__ char smem[];
  unsigned short* yB   = (unsigned short*)smem;                       // 128 x STR bf16 (y, later att)
  unsigned short* attB = yB;                                          // alias: reused after GEMM1
  unsigned short* yhB  = (unsigned short*)(smem + (size_t)LSEQ * STR * 2);
  float* fsc  = (float*)(smem + (size_t)2 * LSEQ * STR * 2);
  float* xL   = fsc;         // 512
  float* xhL  = fsc + 512;
  float* qL   = fsc + 1024;  // query
  float* izL  = fsc + 1536;  // 1/Z per column
  float* hsL  = fsc + 2048;  // h accumulator
  float* fcbL = fsc + 2560;
  float* qbL  = fsc + 3072;
  float* kbL  = fsc + 3584;
  float* vbL  = fsc + 4096;

  const unsigned short* fcwB = wbf;
  const unsigned short* kwB  = wbf + WMAT;
  const unsigned short* vwB  = wbf + 2 * WMAT;

  const int tid  = threadIdx.x;
  const int lane = tid & 31;      // wave32
  const int wv   = tid >> 5;      // wave 0..7 -> owns private 64-column strip
  const int b    = blockIdx.x;

  // ---- phase 0: biases + x to LDS; y[b] fp32 -> bf16 LDS (read y exactly once) ----
  for (int i = tid; i < DIM; i += 256) {
    xL[i]   = x[(size_t)b * DIM + i];
    fcbL[i] = fc_b[i];
    qbL[i]  = q_b[i];
    kbL[i]  = k_b[i];
    vbL[i]  = v_b[i];
  }
  const float* yb = y + (size_t)b * LSEQ * DIM;
  for (int i = tid; i < LSEQ * (DIM / 8); i += 256) {
    int row = i >> 6, g = i & 63;
    union { float4 q4[2]; f32x8 v; } raw;
    const float4* p = (const float4*)(yb + (size_t)row * DIM + g * 8);
    raw.q4[0] = p[0]; raw.q4[1] = p[1];
    Pack8 pk; pk.v = __builtin_convertvector(raw.v, bf16x8);
    *(uint4*)(yB + (size_t)row * STR + g * 8) = pk.u4;
  }
  __syncthreads();

  // ---- xh GEMV: xh = lrelu(x . fc_w^T + fc_b) ----
  for (int h = tid; h < DIM; h += 256) {
    const float4* wr = (const float4*)(fc_w + (size_t)h * DIM);
    float s = fcbL[h];
    for (int d4 = 0; d4 < DIM / 4; ++d4) {
      float4 w4 = wr[d4];
      int d = d4 * 4;
      s += xL[d] * w4.x + xL[d + 1] * w4.y + xL[d + 2] * w4.z + xL[d + 3] * w4.w;
    }
    xhL[h] = lrelu(s);
  }

  // ---- GEMM1: yh = lrelu(y . fc_w^T + fc_b) -> bf16 LDS ----
  {
    f32x8 acc[4][8] = {};
    gemm_strip4(yB, fcwB, wv * 64, lane, acc);
#pragma unroll
    for (int s = 0; s < 4; ++s) epi_yh(acc[s], yhB, fcbL, wv * 64 + 16 * s, lane);
  }
  __syncthreads();   // yhB complete; yB region now dead -> reusable as attB

  // ---- query GEMV: q = xh . q_w^T + q_b ; zero h accumulator ----
  for (int h = tid; h < DIM; h += 256) {
    const float4* wr = (const float4*)(q_w + (size_t)h * DIM);
    float s = qbL[h];
    for (int d4 = 0; d4 < DIM / 4; ++d4) {
      float4 w4 = wr[d4];
      int d = d4 * 4;
      s += xhL[d] * w4.x + xhL[d + 1] * w4.y + xhL[d + 2] * w4.z + xhL[d + 3] * w4.w;
    }
    qL[h] = s;
    hsL[h] = 0.f;
  }
  __syncthreads();

  // ---- GEMM2: att = (yh . k_w^T + k_b) * query -> bf16 LDS (reuses y space) ----
  {
    f32x8 acc[4][8] = {};
    gemm_strip4(yhB, kwB, wv * 64, lane, acc);
#pragma unroll
    for (int s = 0; s < 4; ++s) epi_att(acc[s], attB, kbL, qL, wv * 64 + 16 * s, lane);
  }
  __syncthreads();

  // ---- softmax over L per feature column (2 columns per thread) ----
  for (int h = tid; h < DIM; h += 256) {
    float mx = -3.0e38f;
    for (int l = 0; l < LSEQ; ++l) mx = fmaxf(mx, bf2f(attB[(size_t)l * STR + h]));
    float Z = 0.f;
    for (int l = 0; l < LSEQ; ++l) {
      float e = __expf(bf2f(attB[(size_t)l * STR + h]) - mx);
      Z += e;
      attB[(size_t)l * STR + h] = f2bfh(e);   // store unnormalized exp
    }
    izL[h] = 1.0f / Z;
  }
  __syncthreads();

  // ---- GEMM3: vec = yh . v_w^T + v_b ; h += sum_l lrelu(vec * e) (LDS atomic reduce) ----
  {
    f32x8 acc[4][8] = {};
    gemm_strip4(yhB, vwB, wv * 64, lane, acc);
#pragma unroll
    for (int s = 0; s < 4; ++s) epi_vec(acc[s], attB, vbL, izL, hsL, wv * 64 + 16 * s, lane);
  }
  __syncthreads();

  // ---- out = (xh + h) / 2 ----
  for (int h = tid; h < DIM; h += 256)
    out[(size_t)b * DIM + h] = 0.5f * (xhL[h] + hsL[h]);
}

extern "C" void kernel_launch(void* const* d_in, const int* in_sizes, int n_in,
                              void* d_out, int out_size, void* d_ws, size_t ws_size,
                              hipStream_t stream) {
  (void)in_sizes; (void)n_in; (void)out_size; (void)ws_size;
  const float* x   = (const float*)d_in[0];
  const float* y   = (const float*)d_in[1];
  const float* fcw = (const float*)d_in[2];
  const float* fcb = (const float*)d_in[3];
  const float* qw  = (const float*)d_in[4];
  const float* qb  = (const float*)d_in[5];
  const float* kw  = (const float*)d_in[6];
  const float* kb  = (const float*)d_in[7];
  const float* vw  = (const float*)d_in[8];
  const float* vb  = (const float*)d_in[9];
  float* out = (float*)d_out;
  unsigned short* wbf = (unsigned short*)d_ws;   // 3 * 512*512 bf16 = 1.5 MB

  // pre-pass: convert the three GEMM weight matrices to bf16 (deterministic, every call)
  cvt_weights_kernel<<<dim3(WMAT / (256 * 8)), dim3(256), 0, stream>>>(fcw, wbf);
  cvt_weights_kernel<<<dim3(WMAT / (256 * 8)), dim3(256), 0, stream>>>(kw,  wbf + WMAT);
  cvt_weights_kernel<<<dim3(WMAT / (256 * 8)), dim3(256), 0, stream>>>(vw,  wbf + 2 * WMAT);

  const size_t SMEM = (size_t)2 * LSEQ * STR * 2 + 4608 * 4;  // 284,672 B (<320 KB WGP LDS)
  (void)hipFuncSetAttribute((const void*)ext_attn_kernel,
                            hipFuncAttributeMaxDynamicSharedMemorySize, (int)SMEM);
  ext_attn_kernel<<<dim3(BATCH), dim3(256), SMEM, stream>>>(
      x, y, fcw, fcb, qw, qb, kb, vb, wbf, out);
}
